// CRF_13786845020521
// MI455X (gfx1250) — matrix operand order repaired and checked by reading, not compiled, not run on hardware
//
#include <hip/hip_runtime.h>
#include <hip/hip_bf16.h>

typedef __attribute__((ext_vector_type(16))) _Float16 v16h;
typedef __attribute__((ext_vector_type(8)))  float    v8f;

#define T_N       512
#define L_N       64
#define START_TAG 62
#define STOP_TAG  63

// reduce across the 16 lanes of each half-wave (C-tile column group)
__device__ __forceinline__ float group16_max(float x) {
#pragma unroll
  for (int m = 1; m < 16; m <<= 1) x = fmaxf(x, __shfl_xor(x, m, 32));
  return x;
}
__device__ __forceinline__ float group16_sum(float x) {
#pragma unroll
  for (int m = 1; m < 16; m <<= 1) x += __shfl_xor(x, m, 32);
  return x;
}

// One wave = 16 batch rows, full T scan.
// State: f[b,j] = m[b] + log(s[b,j]); s kept in the 4 WMMA C-tiles (f32),
// m[b] kept as an integer count of power-of-two rescales (exact).
__global__ __launch_bounds__(32) void crf_fwd_wmma(
    const float* __restrict__ em,   // [B, T, L] emissions
    const float* __restrict__ tr,   // [L, L] transitions[next, prev]
    float* __restrict__ out)        // [B]
{
  __shared__ _Float16 esh[16 * 66];          // 16 rows x 64 halves, stride 66 (33 dwords -> conflict-free)

  const int lane = threadIdx.x & 31;
  const int g    = lane >> 4;                // half-wave group
  const int ln   = lane & 15;
  const int bb   = blockIdx.x * 16;          // first batch row of this wave

  // ---- constant B operand: E[K=prev, N=next] = exp(trans[next, prev]) ----
  // B operand (32x16, 16-bit): lane holds column N = lane%16; half h -> K = g*16 + h (contiguous k order, same as A).
  v16h bf[4][2];
#pragma unroll
  for (int n0 = 0; n0 < 4; ++n0) {
    const int N = n0 * 16 + ln;
#pragma unroll
    for (int k0 = 0; k0 < 2; ++k0) {
      v16h f;
#pragma unroll
      for (int h = 0; h < 16; ++h) {
        const int K = k0 * 32 + g * 16 + h;
        f[h] = (_Float16)__expf(tr[N * L_N + K]);
      }
      bf[n0][k0] = f;
    }
  }

  // ---- init: f = NEG everywhere except f[:,START] = 0  ->  s = delta(START), m = 0 ----
  v8f s[4];
#pragma unroll
  for (int n0 = 0; n0 < 4; ++n0)
#pragma unroll
    for (int r = 0; r < 8; ++r) s[n0][r] = 0.0f;
  if (ln == (START_TAG & 15)) {
#pragma unroll
    for (int r = 0; r < 8; ++r) s[START_TAG >> 4][r] = 1.0f;
  }
  int kacc[8];
#pragma unroll
  for (int r = 0; r < 8; ++r) kacc[r] = 0;

  const float* emb = em + (size_t)bb * T_N * L_N;

  for (int t = 0; t < T_N; ++t) {
    // ---- C-layout (M = r + 8g, N = tile*16 + ln) -> LDS row-major f16 tile ----
#pragma unroll
    for (int r = 0; r < 8; ++r) {
      const int row = r + 8 * g;
#pragma unroll
      for (int n0 = 0; n0 < 4; ++n0)
        esh[row * 66 + n0 * 16 + ln] = (_Float16)s[n0][r];
    }
    // single wave: DS ops are in-order; hard fence anyway (CDNA5 split counter)
    asm volatile("s_wait_dscnt 0" ::: "memory");

    // ---- A operand (16x32 f16): lane holds row M = lane%16; half h -> K = g*16 + h ----
    v16h a0, a1;
#pragma unroll
    for (int h = 0; h < 16; ++h) {
      const int v = h >> 1, sb = h & 1;
      const int K = g * 16 + 2 * v + sb;
      a0[h] = esh[ln * 66 + K];
      a1[h] = esh[ln * 66 + 32 + K];
    }

    // ---- emissions for step t (coalesced 64B per half-wave row), exp'd ----
    float eem[4][8];
#pragma unroll
    for (int r = 0; r < 8; ++r) {
      const int row = r + 8 * g;
      const size_t base = ((size_t)row * T_N + t) * L_N;
#pragma unroll
      for (int n0 = 0; n0 < 4; ++n0)
        eem[n0][r] = __expf(emb[base + n0 * 16 + ln]);
    }
    if (t + 4 < T_N) {
#pragma unroll
      for (int r = 0; r < 8; ++r) {
        const int row = r + 8 * g;
        __builtin_prefetch(&emb[((size_t)row * T_N + (t + 4)) * L_N + ln * 4], 0, 0);
      }
    }

    // ---- acc[M,N] = sum_prev s[M,prev] * exp(trans[N,prev])  (8x WMMA f16->f32) ----
    v8f acc[4];
#pragma unroll
    for (int n0 = 0; n0 < 4; ++n0) {
      v8f z = {};
      z = __builtin_amdgcn_wmma_f32_16x16x32_f16(false, a0, false, bf[n0][0],
                                                 (short)0, z, false, false);
      z = __builtin_amdgcn_wmma_f32_16x16x32_f16(false, a1, false, bf[n0][1],
                                                 (short)0, z, false, false);
      acc[n0] = z;
    }

    // ---- u = acc * exp(emit); rescale by exact 2^-k (k = exponent of row max) ----
#pragma unroll
    for (int r = 0; r < 8; ++r) {
      float u0 = acc[0][r] * eem[0][r];
      float u1 = acc[1][r] * eem[1][r];
      float u2 = acc[2][r] * eem[2][r];
      float u3 = acc[3][r] * eem[3][r];
      float mx = fmaxf(fmaxf(u0, u1), fmaxf(u2, u3));
      mx = group16_max(mx);
      mx = fmaxf(mx, 1e-30f);
      const int k = (__float_as_int(mx) >> 23) - 127;
      kacc[r] += k;
      const float sc = __int_as_float((127 - k) << 23);   // exact 2^-k
      s[0][r] = u0 * sc; s[1][r] = u1 * sc;
      s[2][r] = u2 * sc; s[3][r] = u3 * sc;
    }
  }

  // ---- terminal: out[b] = m[b] + log( sum_j s[b,j] * exp(trans[STOP, j]) ) ----
  float est[4];
#pragma unroll
  for (int n0 = 0; n0 < 4; ++n0)
    est[n0] = __expf(tr[STOP_TAG * L_N + n0 * 16 + ln]);
#pragma unroll
  for (int r = 0; r < 8; ++r) {
    float w = s[0][r] * est[0] + s[1][r] * est[1] +
              s[2][r] * est[2] + s[3][r] * est[3];
    w = group16_sum(w);
    if (ln == 0) {
      const int row = r + 8 * g;
      out[bb + row] = (float)kacc[r] * 0.6931471805599453f +
                      __logf(fmaxf(w, 1e-37f));
    }
  }
}

extern "C" void kernel_launch(void* const* d_in, const int* in_sizes, int n_in,
                              void* d_out, int out_size, void* d_ws, size_t ws_size,
                              hipStream_t stream) {
  const float* em = (const float*)d_in[0];   // [B, T, L] float32
  const float* tr = (const float*)d_in[1];   // [L, L]    float32
  float* out = (float*)d_out;                // [B]       float32

  const int B = in_sizes[0] / (T_N * L_N);   // 512 for the reference setup
  dim3 grid(B / 16), block(32);              // one wave per 16 batch rows
  crf_fwd_wmma<<<grid, block, 0, stream>>>(em, tr, out);
}
